// RealFormer_30691836297553
// MI455X (gfx1250) — matrix operand order repaired and checked by reading, not compiled, not run on hardware
//
#include <hip/hip_runtime.h>
#include <math.h>

// ---------------------------------------------------------------------------
// RealFormer forward for MI455X (gfx1250): bf16 WMMA GEMMs + f32 softmax/LN.
// B=8, T=512, H=8, ES=96, EMB=768, L=6.
// Staging uses GLOBAL_LOAD_ASYNC_TO_LDS_B128 (ASYNCcnt) instead of
// load->wait->ds_store round-trips.
// ---------------------------------------------------------------------------

typedef __attribute__((ext_vector_type(16))) __bf16 v16bf;
typedef __attribute__((ext_vector_type(8)))  __bf16 v8bf;
typedef __attribute__((ext_vector_type(8)))  float  v8f;

union Frag16 { v16bf v; v8bf h[2]; };

#define GM_KQV    0
#define GM_SCORES 1
#define GM_AV     2
#define GM_PROJ   3
#define GM_FFN1   4
#define GM_FFN2   5

// Tile config: 256 threads = 8 waves (wave32). Wave grid 2(M) x 4(N).
// Each wave owns 64x32 = 4x2 tiles of 16x16. Block tile 128x128, K step 32.
#define BM 128
#define BN 128
#define BK 32
#define LDSK 40   // 80B row stride: 16B aligned, reduces bank conflicts

__device__ __forceinline__ void async_copy_b128(unsigned ldsOff, const void* gaddr) {
  // GLOBAL_LOAD_ASYNC_TO_LDS_B128: per-lane 16B global -> LDS, tracked by ASYNCcnt.
  asm volatile("global_load_async_to_lds_b128 %0, %1, off"
               :: "v"(ldsOff), "v"(gaddr) : "memory");
}
__device__ __forceinline__ void wait_async0() {
  asm volatile("s_wait_asynccnt 0x0" ::: "memory");
}

struct GemmParams {
  const __bf16* A;          // A[m*ldA + k], k contiguous
  const __bf16* B;          // B[n*ldBn + k*ldBk]  (i.e. W[n][k] style)
  long long ldA, ldBn, ldBk;
  long long bA_o, bA_i;     // batch base = (z>>3)*o + (z&7)*i
  long long bB_o, bB_i;
  long long bC_o, bC_i;
  int M, N, K;
  float alpha;
  long long ldC;
  float*  outF;             // f32 output (SCORES / PROJ / FFN2)
  __bf16* outB;             // bf16 output (AV / FFN1)
  const float* add;         // prev scores or residual (same layout as C)
  const float* bias;        // per-n bias (FFN1 / FFN2)
  __bf16* outK; __bf16* outQ; __bf16* outV;   // KQV split outputs
};

template <int MODE>
__global__ __launch_bounds__(256)
void gemm_bf16_kernel(GemmParams p) {
  __shared__ __bf16 As[BM * LDSK];
  __shared__ __bf16 Bs[BN * LDSK];

  const int tid  = threadIdx.x;
  const int lane = tid & 31;
  const int wid  = tid >> 5;
  const int wm   = wid & 1;      // wave row (M)
  const int wn   = wid >> 1;     // wave col (N)
  const int l16  = lane & 15;
  const int half = lane >> 4;

  const int z  = blockIdx.z;
  const int zo = z >> 3, zi = z & 7;
  const __bf16* Ab = p.A + (long long)zo * p.bA_o + (long long)zi * p.bA_i;
  const __bf16* Bb = p.B + (long long)zo * p.bB_o + (long long)zi * p.bB_i;
  const long long cbase = (long long)zo * p.bC_o + (long long)zi * p.bC_i;

  const int blockM = blockIdx.y * BM;
  const int blockN = blockIdx.x * BN;

  // N-guard rows of Bs are K-invariant: zero them once, never touch again.
  if (blockN + BN > p.N) {
#pragma unroll
    for (int r = 0; r < 2; ++r) {
      const int g  = tid + r * 256;
      const int n  = g >> 2;
      const int kk = (g & 3) << 3;
      if (blockN + n >= p.N) {
        v8bf z8 = {(__bf16)0.f,(__bf16)0.f,(__bf16)0.f,(__bf16)0.f,
                   (__bf16)0.f,(__bf16)0.f,(__bf16)0.f,(__bf16)0.f};
        *(v8bf*)&Bs[n * LDSK + kk] = z8;
      }
    }
  }

  v8f acc[4][2] = {};

  for (int k0 = 0; k0 < p.K; k0 += BK) {
    __syncthreads();   // prior iteration finished reading LDS
    // ---- stage A tile: BM x BK, async DMA, 8 bf16 (16B) per lane-op ----
#pragma unroll
    for (int r = 0; r < 2; ++r) {
      const int g  = tid + r * 256;          // 512 groups of 8
      const int m  = g >> 2;
      const int kk = (g & 3) << 3;
      const __bf16* src = Ab + (long long)(blockM + m) * p.ldA + k0 + kk;
      async_copy_b128((unsigned)(size_t)&As[m * LDSK + kk], src);
      if (k0 + BK < p.K) __builtin_prefetch(src + BK, 0, 3);  // next K tile -> near caches
    }
    // ---- stage B tile: BN rows (n) x BK (k) ----
#pragma unroll
    for (int r = 0; r < 2; ++r) {
      const int g  = tid + r * 256;
      const int n  = g >> 2;
      const int kk = (g & 3) << 3;
      const int gn = blockN + n;
      if (gn < p.N) {
        if (p.ldBk == 1) {
          const __bf16* src = Bb + (long long)gn * p.ldBn + k0 + kk;
          async_copy_b128((unsigned)(size_t)&Bs[n * LDSK + kk], src);
        } else {
          v8bf bv;
#pragma unroll
          for (int j = 0; j < 8; ++j)
            bv[j] = Bb[(long long)gn * p.ldBn + (long long)(k0 + kk + j) * p.ldBk];
          *(v8bf*)&Bs[n * LDSK + kk] = bv;
        }
      }
    }
    wait_async0();     // all async LDS writes from this wave landed
    __syncthreads();   // ... and from every other wave

    // ---- fragments (ISA 7.12.2 16-bit layouts) ----
    Frag16 afr[4], bfr[2];
#pragma unroll
    for (int i = 0; i < 4; ++i) {
      // A: lane l holds row M=l%16; half 0 -> K 0..7 & 16..23, half 1 -> +8
      const __bf16* pa = &As[(wm * 64 + i * 16 + l16) * LDSK + 8 * half];
      afr[i].h[0] = *(const v8bf*)(pa);        // K = 8h .. 8h+7
      afr[i].h[1] = *(const v8bf*)(pa + 16);   // K = 16+8h .. 23+8h
    }
#pragma unroll
    for (int j = 0; j < 2; ++j) {
      // B: lane l holds col N=l%16; half 0 -> K 0..7 & 16..23, half 1 -> +8
      const __bf16* pb = &Bs[(wn * 32 + j * 16 + l16) * LDSK + 8 * half];
      bfr[j].h[0] = *(const v8bf*)(pb);
      bfr[j].h[1] = *(const v8bf*)(pb + 16);
    }
#pragma unroll
    for (int i = 0; i < 4; ++i)
#pragma unroll
      for (int j = 0; j < 2; ++j)
        acc[i][j] = __builtin_amdgcn_wmma_f32_16x16x32_bf16(
            false, afr[i].v, false, bfr[j].v, (short)0, acc[i][j], false, false);
  }

  // ---- epilogue: C layout VGPR r -> M = base + r + 8*half, N = base + l%16 ----
#pragma unroll
  for (int i = 0; i < 4; ++i) {
#pragma unroll
    for (int j = 0; j < 2; ++j) {
      const int n = blockN + wn * 32 + j * 16 + l16;
      if (n >= p.N) continue;
      const int mbase = blockM + wm * 64 + i * 16 + half * 8;
#pragma unroll
      for (int r = 0; r < 8; ++r) {
        const int m = mbase + r;
        const float v = p.alpha * acc[i][j][r];
        if constexpr (MODE == GM_KQV) {
          const int sel = (n >= 192) ? 2 : (n >= 96 ? 1 : 0);
          const int nn  = n - sel * 96;
          __bf16* dst = (sel == 0) ? p.outK : (sel == 1 ? p.outQ : p.outV);
          dst[(long long)m * 96 + nn] = (__bf16)v;
        } else if constexpr (MODE == GM_SCORES) {
          const long long idx = cbase + (long long)m * p.ldC + n;
          p.outF[idx] = v + p.add[idx];               // raw score + prev layer score
        } else if constexpr (MODE == GM_AV) {
          p.outB[cbase + (long long)m * p.ldC + n] = (__bf16)v;
        } else if constexpr (MODE == GM_PROJ) {
          const long long idx = (long long)m * p.ldC + n;
          p.outF[idx] = v + p.add[idx];               // + residual x
        } else if constexpr (MODE == GM_FFN1) {
          const float t = v + p.bias[n];
          const float ge = 0.5f * t * (1.0f + erff(t * 0.70710678118654752f));
          p.outB[(long long)m * p.ldC + n] = (__bf16)ge;
        } else {                                       // GM_FFN2
          const long long idx = (long long)m * p.ldC + n;
          p.outF[idx] = v + p.bias[n] + p.add[idx];
        }
      }
    }
  }
}

// ---------------- softmax over key axis (rows of 512) ----------------------
__global__ __launch_bounds__(256)
void softmax_kernel(const float* __restrict__ S, __bf16* __restrict__ P) {
  const long long row = blockIdx.x;
  const float* s = S + row * 512;
  __bf16*      p = P + row * 512;
  __shared__ float red[256];
  const int tid = threadIdx.x;
  const float a = s[tid], b = s[tid + 256];
  red[tid] = fmaxf(a, b);
  __syncthreads();
  for (int off = 128; off > 0; off >>= 1) {
    if (tid < off) red[tid] = fmaxf(red[tid], red[tid + off]);
    __syncthreads();
  }
  const float rmax = red[0];
  __syncthreads();
  const float e0 = __expf(a - rmax), e1 = __expf(b - rmax);
  red[tid] = e0 + e1;
  __syncthreads();
  for (int off = 128; off > 0; off >>= 1) {
    if (tid < off) red[tid] += red[tid + off];
    __syncthreads();
  }
  const float inv = 1.0f / red[0];
  p[tid]       = (__bf16)(e0 * inv);
  p[tid + 256] = (__bf16)(e1 * inv);
}

// ---------------- layernorm over rows of 768 -------------------------------
__global__ __launch_bounds__(256)
void layernorm_kernel(const float* __restrict__ in, const float* __restrict__ g,
                      const float* __restrict__ b, float* __restrict__ outF,
                      __bf16* __restrict__ outB) {
  const long long row = blockIdx.x;
  const int tid = threadIdx.x;
  const float* x = in + row * 768;
  const float v0 = x[tid], v1 = x[tid + 256], v2 = x[tid + 512];
  __shared__ float red[256];
  red[tid] = v0 + v1 + v2;
  __syncthreads();
  for (int off = 128; off > 0; off >>= 1) {
    if (tid < off) red[tid] += red[tid + off];
    __syncthreads();
  }
  const float mu = red[0] * (1.0f / 768.0f);
  __syncthreads();
  const float d0 = v0 - mu, d1 = v1 - mu, d2 = v2 - mu;
  red[tid] = d0 * d0 + d1 * d1 + d2 * d2;
  __syncthreads();
  for (int off = 128; off > 0; off >>= 1) {
    if (tid < off) red[tid] += red[tid + off];
    __syncthreads();
  }
  const float inv = rsqrtf(red[0] * (1.0f / 768.0f) + 1e-5f);
  const float d[3] = {d0, d1, d2};
#pragma unroll
  for (int e = 0; e < 3; ++e) {
    const int col = tid + e * 256;
    const float o = d[e] * inv * g[col] + b[col];
    outF[row * 768 + col] = o;
    outB[row * 768 + col] = (__bf16)o;
  }
}

// ---------------- utility kernels ------------------------------------------
__global__ void convert_bf16_kernel(const float* __restrict__ in,
                                    __bf16* __restrict__ out, long long n) {
  long long i = (long long)blockIdx.x * blockDim.x + threadIdx.x;
  const long long stride = (long long)gridDim.x * blockDim.x;
  for (; i < n; i += stride) out[i] = (__bf16)in[i];
}

__global__ void copy_x_kernel(const float* __restrict__ in, float* __restrict__ xf,
                              __bf16* __restrict__ xb, long long n) {
  long long i = (long long)blockIdx.x * blockDim.x + threadIdx.x;
  const long long stride = (long long)gridDim.x * blockDim.x;
  for (; i < n; i += stride) { const float v = in[i]; xf[i] = v; xb[i] = (__bf16)v; }
}

__global__ void zero_f32_kernel(float* __restrict__ out, long long n) {
  long long i = (long long)blockIdx.x * blockDim.x + threadIdx.x;
  const long long stride = (long long)gridDim.x * blockDim.x;
  for (; i < n; i += stride) out[i] = 0.0f;
}

// ---------------------------------------------------------------------------
extern "C" void kernel_launch(void* const* d_in, const int* in_sizes, int n_in,
                              void* d_out, int out_size, void* d_ws, size_t ws_size,
                              hipStream_t stream) {
  const float* x_in  = (const float*)d_in[0];
  const float* Wkqv  = (const float*)d_in[1];
  const float* Wproj = (const float*)d_in[2];
  const float* ln1_g = (const float*)d_in[3];
  const float* ln1_b = (const float*)d_in[4];
  const float* ln2_g = (const float*)d_in[5];
  const float* ln2_b = (const float*)d_in[6];
  const float* ff_w1 = (const float*)d_in[7];
  const float* ff_b1 = (const float*)d_in[8];
  const float* ff_w2 = (const float*)d_in[9];
  const float* ff_b2 = (const float*)d_in[10];

  const int L = 6, T = 512, Bn = 8, H = 8, ES = 96, EMB = 768;
  const long long NTOK = (long long)Bn * T;           // 4096
  const long long wkqvL = 3LL * ES * ES;              // 27648
  const long long wprojL = (long long)EMB * EMB;      // 589824
  const long long w1L = 4LL * EMB * EMB;              // 2359296
  const long long w2L = 4LL * EMB * EMB;
  const long long scoreE = (long long)Bn * H * T * T; // 16,777,216

  char* ws = (char*)d_ws;
  size_t off = 0;
  auto alloc = [&](size_t bytes) -> void* {
    void* p = ws + off;
    off = (off + bytes + 255) & ~(size_t)255;
    return p;
  };
  __bf16* wkqv_b  = (__bf16*)alloc(L * wkqvL * 2);
  __bf16* wproj_b = (__bf16*)alloc(L * wprojL * 2);
  __bf16* w1_b    = (__bf16*)alloc(L * w1L * 2);
  __bf16* w2_b    = (__bf16*)alloc(L * w2L * 2);
  float*  x_f32   = (float*) alloc(NTOK * EMB * 4);
  __bf16* xb      = (__bf16*)alloc(NTOK * EMB * 2);
  __bf16* kb      = (__bf16*)alloc(NTOK * EMB * 2);
  __bf16* qb      = (__bf16*)alloc(NTOK * EMB * 2);
  __bf16* vb      = (__bf16*)alloc(NTOK * EMB * 2);
  float*  prev    = (float*) alloc(scoreE * 4);
  float*  cur     = (float*) alloc(scoreE * 4);
  __bf16* att_b   = (__bf16*)alloc(scoreE * 2);
  __bf16* attnb   = (__bf16*)alloc(NTOK * EMB * 2);
  float*  tmp     = (float*) alloc(NTOK * EMB * 4);
  __bf16* hb      = (__bf16*)alloc(NTOK * 4LL * EMB * 2);

  // weights -> bf16
  convert_bf16_kernel<<<512, 256, 0, stream>>>(Wkqv,  wkqv_b,  L * wkqvL);
  convert_bf16_kernel<<<1024, 256, 0, stream>>>(Wproj, wproj_b, L * wprojL);
  convert_bf16_kernel<<<2048, 256, 0, stream>>>(ff_w1, w1_b,    L * w1L);
  convert_bf16_kernel<<<2048, 256, 0, stream>>>(ff_w2, w2_b,    L * w2L);
  // x -> f32 state + bf16 activation; prev scores = 0
  copy_x_kernel<<<1024, 256, 0, stream>>>(x_in, x_f32, xb, NTOK * EMB);
  zero_f32_kernel<<<2048, 256, 0, stream>>>(prev, scoreE);

  const float scale = 0.10206207261596575f;  // 1/sqrt(96)

  for (int l = 0; l < L; ++l) {
    GemmParams p{};
    // ---- 1) KQV: [32768,96] x Wkqv^T[96,288] -> split k,q,v (bf16) ----
    p = GemmParams{};
    p.A = xb;  p.ldA = ES;                       // view x as [B*T*H, 96]
    p.B = wkqv_b + (long long)l * wkqvL; p.ldBn = ES; p.ldBk = 1;
    p.M = (int)(NTOK * H); p.N = 3 * ES; p.K = ES;
    p.alpha = 1.0f; p.outK = kb; p.outQ = qb; p.outV = vb;
    gemm_bf16_kernel<GM_KQV><<<dim3(3, 256, 1), 256, 0, stream>>>(p);

    // ---- 2) scores: per (b,h): q[512,96] x k^T -> cur = s*qk + prev ----
    p = GemmParams{};
    p.A = qb; p.ldA = EMB; p.bA_o = (long long)T * EMB; p.bA_i = ES;
    p.B = kb; p.ldBn = EMB; p.ldBk = 1; p.bB_o = (long long)T * EMB; p.bB_i = ES;
    p.M = T; p.N = T; p.K = ES; p.alpha = scale;
    p.ldC = T; p.bC_o = 8LL * T * T; p.bC_i = (long long)T * T;
    p.outF = cur; p.add = prev;
    gemm_bf16_kernel<GM_SCORES><<<dim3(4, 4, Bn * H), 256, 0, stream>>>(p);

    // ---- 3) softmax over keys -> bf16 probabilities ----
    softmax_kernel<<<dim3((unsigned)(Bn * H * T)), 256, 0, stream>>>(cur, att_b);

    // ---- 4) att x v -> attn output (bf16, token-major [4096,768]) ----
    p = GemmParams{};
    p.A = att_b; p.ldA = T; p.bA_o = 8LL * T * T; p.bA_i = (long long)T * T;
    p.B = vb; p.ldBn = 1; p.ldBk = EMB; p.bB_o = (long long)T * EMB; p.bB_i = ES;
    p.M = T; p.N = ES; p.K = T; p.alpha = 1.0f;
    p.ldC = EMB; p.bC_o = (long long)T * EMB; p.bC_i = ES;
    p.outB = attnb;
    gemm_bf16_kernel<GM_AV><<<dim3(1, 4, Bn * H), 256, 0, stream>>>(p);

    // ---- 5) output projection + residual ----
    p = GemmParams{};
    p.A = attnb; p.ldA = EMB;
    p.B = wproj_b + (long long)l * wprojL; p.ldBn = EMB; p.ldBk = 1;
    p.M = (int)NTOK; p.N = EMB; p.K = EMB; p.alpha = 1.0f;
    p.ldC = EMB; p.outF = tmp; p.add = x_f32;
    gemm_bf16_kernel<GM_PROJ><<<dim3(6, 32, 1), 256, 0, stream>>>(p);

    // ---- 6) LN1 -> x (f32) + xb (bf16) ----
    layernorm_kernel<<<dim3((unsigned)NTOK), 256, 0, stream>>>(
        tmp, ln1_g + (long long)l * EMB, ln1_b + (long long)l * EMB, x_f32, xb);

    // ---- 7) FFN1 + GELU -> hb (bf16) ----
    p = GemmParams{};
    p.A = xb; p.ldA = EMB;
    p.B = w1_b + (long long)l * w1L; p.ldBn = EMB; p.ldBk = 1;
    p.M = (int)NTOK; p.N = 4 * EMB; p.K = EMB; p.alpha = 1.0f;
    p.ldC = 4 * EMB; p.outB = hb; p.bias = ff_b1 + (long long)l * 4 * EMB;
    gemm_bf16_kernel<GM_FFN1><<<dim3(24, 32, 1), 256, 0, stream>>>(p);

    // ---- 8) FFN2 + bias + residual ----
    p = GemmParams{};
    p.A = hb; p.ldA = 4 * EMB;
    p.B = w2_b + (long long)l * w2L; p.ldBn = 4 * EMB; p.ldBk = 1;
    p.M = (int)NTOK; p.N = EMB; p.K = 4 * EMB; p.alpha = 1.0f;
    p.ldC = EMB; p.outF = tmp; p.add = x_f32; p.bias = ff_b2 + (long long)l * EMB;
    gemm_bf16_kernel<GM_FFN2><<<dim3(6, 32, 1), 256, 0, stream>>>(p);

    // ---- 9) LN2 -> next x; final layer writes d_out directly ----
    float* lnOut = (l == L - 1) ? (float*)d_out : x_f32;
    layernorm_kernel<<<dim3((unsigned)NTOK), 256, 0, stream>>>(
        tmp, ln2_g + (long long)l * EMB, ln2_b + (long long)l * EMB, lnOut, xb);

    // residual score carry
    float* t2 = prev; prev = cur; cur = t2;
  }
}
